// ExpertFFNEnsemble_63513976373279
// MI455X (gfx1250) — compile-verified, matrix operand order checked by reading, not run on hardware
//
#include <hip/hip_runtime.h>
#include <math.h>

// ---------------- types ----------------
typedef __attribute__((ext_vector_type(16))) __bf16 v16bf;
typedef __attribute__((ext_vector_type(8)))  __bf16 v8bf;
typedef __attribute__((ext_vector_type(8)))  float  v8f;
typedef __attribute__((ext_vector_type(4)))  int    v4i;

// ------------- optional CDNA5 async global->LDS path -------------
#if defined(__has_builtin)
# if __has_builtin(__builtin_amdgcn_global_load_async_to_lds_b128)
#  define HAVE_ASYNC_LDS 1
# endif
#endif

#ifdef HAVE_ASYNC_LDS
typedef __attribute__((address_space(1))) v4i* gv4i_p;   // global v4i*
typedef __attribute__((address_space(3))) v4i* lv4i_p;   // LDS v4i*
__device__ __forceinline__ void async_cp16(void* lds_dst, const void* gsrc) {
    // ISA: LDS aperture places the LDS byte offset in addr[31:0]
    lv4i_p l = (lv4i_p)(unsigned int)(size_t)lds_dst;
    gv4i_p g = (gv4i_p)(size_t)gsrc;
    __builtin_amdgcn_global_load_async_to_lds_b128(g, l, 0, 0);
}
__device__ __forceinline__ void wait_async0() {
#if __has_builtin(__builtin_amdgcn_s_wait_asynccnt)
    __builtin_amdgcn_s_wait_asynccnt(0);
#else
    asm volatile("s_wait_asynccnt 0" ::: "memory");
#endif
}
#endif

__device__ __forceinline__ unsigned short f2bf(float f) {
    unsigned int u = __float_as_uint(f);
    unsigned int r = u + 0x7FFFu + ((u >> 16) & 1u);
    return (unsigned short)(r >> 16);
}
__device__ __forceinline__ float gelu_exact(float v) {
    return 0.5f * v * (1.0f + erff(v * 0.70710678118654752440f));
}

// ---------------- elementwise converts ----------------
__global__ __launch_bounds__(256) void cvt_bf16_kernel(const float* __restrict__ src,
                                                       unsigned short* __restrict__ dst, int n) {
    int i = (blockIdx.x * 256 + threadIdx.x) * 4;
    if (i >= n) return;
    float4 v = *(const float4*)(src + i);
    ushort4 o;
    o.x = f2bf(v.x); o.y = f2bf(v.y); o.z = f2bf(v.z); o.w = f2bf(v.w);
    *(ushort4*)(dst + i) = o;
}

__global__ __launch_bounds__(256) void mulvec_kernel(const float* __restrict__ a,
                                                     const float* __restrict__ b,
                                                     float* __restrict__ dst, int n) {
    int i = blockIdx.x * 256 + threadIdx.x;
    if (i < n) dst[i] = a[i] * b[i];
}

// Transpose + convert fp32[z][R][C] (optional per-column scale[z][C]) -> bf16[z][C][R]
__global__ __launch_bounds__(256) void transcvt_kernel(const float* __restrict__ src,
                                                       const float* __restrict__ scale,
                                                       unsigned short* __restrict__ dst,
                                                       int R, int C) {
    __shared__ float t[32][33];
    const int z = blockIdx.z;
    const float* s = src + (size_t)z * R * C;
    unsigned short* d = dst + (size_t)z * R * C;
    const int tx = threadIdx.x, ty = threadIdx.y;
    const int c = blockIdx.x * 32 + tx;
    #pragma unroll
    for (int p = 0; p < 4; ++p) {
        int r = blockIdx.y * 32 + ty + p * 8;
        float v = s[(size_t)r * C + c];
        if (scale) v *= scale[(size_t)z * C + c];
        t[ty + p * 8][tx] = v;
    }
    __syncthreads();
    #pragma unroll
    for (int p = 0; p < 4; ++p) {
        int cc = blockIdx.x * 32 + ty + p * 8;
        int rr = blockIdx.y * 32 + tx;
        d[(size_t)cc * R + rr] = f2bf(t[tx][ty + p * 8]);
    }
}

// ---------------- router: logits -> softmax -> top2 -> renorm ----------------
__global__ __launch_bounds__(256) void router_kernel(const float* __restrict__ x,
                                                     const float* __restrict__ rw,
                                                     const float* __restrict__ rb,
                                                     int* __restrict__ tok_exp,
                                                     float* __restrict__ tok_w,
                                                     int* __restrict__ counts, int N) {
    const int lane = threadIdx.x & 31;
    const int wid = threadIdx.x >> 5;
    const int tok = blockIdx.x * 8 + wid;
    if (tok >= N) return;
    float a[8];
    #pragma unroll
    for (int e = 0; e < 8; ++e) a[e] = 0.f;
    const float* xr = x + (size_t)tok * 1024;
    #pragma unroll 4
    for (int i = 0; i < 32; ++i) {
        int d = i * 32 + lane;
        float xv = xr[d];
        const float4 r0 = *(const float4*)(rw + d * 8);
        const float4 r1 = *(const float4*)(rw + d * 8 + 4);
        a[0] += xv * r0.x; a[1] += xv * r0.y; a[2] += xv * r0.z; a[3] += xv * r0.w;
        a[4] += xv * r1.x; a[5] += xv * r1.y; a[6] += xv * r1.z; a[7] += xv * r1.w;
    }
    #pragma unroll
    for (int off = 16; off > 0; off >>= 1)
        #pragma unroll
        for (int e = 0; e < 8; ++e) a[e] += __shfl_xor(a[e], off, 32);
    if (lane == 0) {
        float l[8];
        #pragma unroll
        for (int e = 0; e < 8; ++e) l[e] = a[e] + rb[e];
        int i0 = 0; float m0 = l[0];
        #pragma unroll
        for (int e = 1; e < 8; ++e) if (l[e] > m0) { m0 = l[e]; i0 = e; }
        int i1 = (i0 == 0) ? 1 : 0; float m1 = l[i1];
        #pragma unroll
        for (int e = 0; e < 8; ++e) if (e != i0 && l[e] > m1) { m1 = l[e]; i1 = e; }
        // renormalized top-2 softmax weights: p0/(p0+p1), p1/(p0+p1)
        float t = expf(m1 - m0);
        float w0 = 1.f / (1.f + t);
        float w1 = t / (1.f + t);
        tok_exp[2 * tok] = i0; tok_exp[2 * tok + 1] = i1;
        tok_w[2 * tok] = w0;  tok_w[2 * tok + 1] = w1;
        atomicAdd(&counts[i0], 1);
        atomicAdd(&counts[i1], 1);
    }
}

__global__ void scan_kernel(const int* __restrict__ counts, int* __restrict__ dbase,
                            int* __restrict__ cursor) {
    if (threadIdx.x == 0 && blockIdx.x == 0) {
        int off = 0;
        for (int e = 0; e < 8; ++e) {
            dbase[e] = off;
            cursor[e] = 0;
            off += ((counts[e] + 127) >> 7) << 7;   // 128-row aligned segments
        }
    }
}

__global__ __launch_bounds__(256) void fill_kernel(const int* __restrict__ tok_exp,
                                                   const int* __restrict__ dbase,
                                                   int* __restrict__ cursor,
                                                   int* __restrict__ rowperm,
                                                   int* __restrict__ tok_row, int N) {
    int tok = blockIdx.x * 256 + threadIdx.x;
    if (tok >= N) return;
    #pragma unroll
    for (int k = 0; k < 2; ++k) {
        int e = tok_exp[2 * tok + k];
        int pos = atomicAdd(&cursor[e], 1);
        int g = dbase[e] + pos;
        rowperm[g] = tok;
        tok_row[2 * tok + k] = g;
    }
}

// ---------------- generic bf16 WMMA GEMM (double-buffered LDS) ----------------
// Out[g, n] = act( A[perm(g), :K] @ W[z][n][:K]^T + bias[z][n] )
// Block tile 128x128, 8 waves (wave tile 64x32), K step 64, v_wmma_f32_16x16x32_bf16.
#define BM 128
#define BN 128
#define BK 64
#define LP 72   // LDS pitch (bf16 elems); 144B rows keep 16B alignment, break conflicts

__global__ __launch_bounds__(256) void gemm_bf16_kernel(
    const unsigned short* __restrict__ A,    // bf16 [rows][K]
    const unsigned short* __restrict__ W,    // bf16 [z][Nout][K]
    const float* __restrict__ bias,          // [z][Nout]
    const int* __restrict__ perm,            // row gather (null -> identity)
    const int* __restrict__ dbase,           // per-z row base (null -> 0)
    const int* __restrict__ dcnt,            // per-z row count (null -> fixedRows)
    int fixedRows, int K, int Nout,
    unsigned short* __restrict__ outB,       // bf16 out (fc1 path) or null
    float* __restrict__ outF,                // f32 out (fc2 path) or null
    int doGelu) {
    __shared__ __align__(16) unsigned short lsA[2][BM * LP];
    __shared__ __align__(16) unsigned short lsB[2][BN * LP];

    const int z = blockIdx.z;
    const int base = dbase ? dbase[z] : 0;
    const int cnt = dcnt ? dcnt[z] : fixedRows;
    const int mt = blockIdx.y;
    if (mt * BM >= cnt) return;                       // uniform early-exit (empty tile)
    int cntLoc = cnt - mt * BM; if (cntLoc > BM) cntLoc = BM;
    const int nb = blockIdx.x * BN;

    const unsigned short* Wz = W + (size_t)z * (size_t)Nout * (size_t)K;
    const float* biasZ = bias + (size_t)z * Nout;

    const int tid = threadIdx.x;
    const int lane = tid & 31;
    const int wid = tid >> 5;
    const int wm = (wid & 1) * 64;                    // wave row offset
    const int wn = (wid >> 1) * 32;                   // wave col offset

    v8f acc[4][2];
    #pragma unroll
    for (int i = 0; i < 4; ++i)
        #pragma unroll
        for (int j = 0; j < 2; ++j)
            #pragma unroll
            for (int r = 0; r < 8; ++r) acc[i][j][r] = 0.f;

    const int lr = tid >> 3;            // 0..31
    const int lc = (tid & 7) * 8;       // 0..56 (16B chunks)

#ifdef HAVE_ASYNC_LDS
    // issue async global->LDS copies for one 128x64 A tile + 128x64 B tile
    auto issue_tile = [&](int k0, int buf) {
        #pragma unroll
        for (int p = 0; p < 4; ++p) {
            int row = p * 32 + lr;
            async_cp16(&lsB[buf][row * LP + lc], Wz + (size_t)(nb + row) * K + k0 + lc);
            if (row < cntLoc) {
                int g = base + mt * BM + row;
                int ar = perm ? perm[g] : g;
                async_cp16(&lsA[buf][row * LP + lc], A + (size_t)ar * K + k0 + lc);
            } else {
                *(uint4*)(&lsA[buf][row * LP + lc]) = make_uint4(0u, 0u, 0u, 0u);
            }
        }
        if (k0 + BK < K)
            __builtin_prefetch(Wz + (size_t)(nb + 2 * lr) * K + k0 + BK + lc, 0, 1);
    };
#else
    uint4 rA[4], rB[4];
    auto fetch = [&](int k0) {                        // global -> regs
        #pragma unroll
        for (int p = 0; p < 4; ++p) {
            int row = p * 32 + lr;
            rB[p] = *(const uint4*)(Wz + (size_t)(nb + row) * K + k0 + lc);
            uint4 va = make_uint4(0u, 0u, 0u, 0u);
            if (row < cntLoc) {
                int g = base + mt * BM + row;
                int ar = perm ? perm[g] : g;
                va = *(const uint4*)(A + (size_t)ar * K + k0 + lc);
            }
            rA[p] = va;
        }
        if (k0 + BK < K)
            __builtin_prefetch(Wz + (size_t)(nb + 2 * lr) * K + k0 + BK + lc, 0, 1);
    };
    auto stash = [&](int buf) {                       // regs -> LDS
        #pragma unroll
        for (int p = 0; p < 4; ++p) {
            int row = p * 32 + lr;
            *(uint4*)(&lsA[buf][row * LP + lc]) = rA[p];
            *(uint4*)(&lsB[buf][row * LP + lc]) = rB[p];
        }
    };
#endif

    auto compute = [&](int buf) {
        const int rsel = (lane >> 4) * 8;             // K-half select per ISA A/B layout
        const int rlo = lane & 15;
        #pragma unroll
        for (int kk = 0; kk < BK; kk += 32) {
            v16bf af[4], bfr[2];
            #pragma unroll
            for (int j = 0; j < 2; ++j) {
                int col = wn + j * 16 + rlo;
                union { v16bf v; v8bf h[2]; } u;
                u.h[0] = *(const v8bf*)(&lsB[buf][col * LP + kk + rsel]);
                u.h[1] = *(const v8bf*)(&lsB[buf][col * LP + kk + 16 + rsel]);
                bfr[j] = u.v;
            }
            #pragma unroll
            for (int i = 0; i < 4; ++i) {
                int row = wm + i * 16 + rlo;
                union { v16bf v; v8bf h[2]; } u;
                u.h[0] = *(const v8bf*)(&lsA[buf][row * LP + kk + rsel]);
                u.h[1] = *(const v8bf*)(&lsA[buf][row * LP + kk + 16 + rsel]);
                af[i] = u.v;
            }
            #pragma unroll
            for (int i = 0; i < 4; ++i)
                #pragma unroll
                for (int j = 0; j < 2; ++j)
                    acc[i][j] = __builtin_amdgcn_wmma_f32_16x16x32_bf16(
                        false, af[i], false, bfr[j], (short)0, acc[i][j], false, false);
        }
    };

    // ---- software pipeline: prefetch tile k+1 while computing tile k ----
    int buf = 0;
#ifdef HAVE_ASYNC_LDS
    issue_tile(0, 0);
    wait_async0();
    __syncthreads();
    for (int k0 = 0; k0 < K; k0 += BK) {
        const bool more = (k0 + BK < K);
        if (more) issue_tile(k0 + BK, buf ^ 1);
        compute(buf);
        if (more) wait_async0();
        __syncthreads();
        buf ^= 1;
    }
#else
    fetch(0);
    stash(0);
    __syncthreads();
    for (int k0 = 0; k0 < K; k0 += BK) {
        const bool more = (k0 + BK < K);
        if (more) fetch(k0 + BK);                     // loads in flight during WMMA
        compute(buf);
        if (more) stash(buf ^ 1);
        __syncthreads();
        buf ^= 1;
    }
#endif

    // Epilogue. C layout: lanes 0-15 -> M = r, N = lane; lanes 16-31 -> M = 8+r, N = lane-16.
    const int rsel = (lane >> 4) * 8;
    const int nlo = lane & 15;
    #pragma unroll
    for (int i = 0; i < 4; ++i) {
        #pragma unroll
        for (int j = 0; j < 2; ++j) {
            int col = nb + wn + j * 16 + nlo;
            float bv = biasZ[col];
            #pragma unroll
            for (int r = 0; r < 8; ++r) {
                int rowLoc = wm + i * 16 + rsel + r;
                if (rowLoc >= cntLoc) continue;
                size_t g = (size_t)(base + mt * BM + rowLoc);
                float v = acc[i][j][r] + bv;
                if (doGelu) v = gelu_exact(v);
                if (outB) outB[g * (size_t)Nout + col] = f2bf(v);
                else      outF[g * (size_t)Nout + col] = v;
            }
        }
    }
}

// ---------------- combine (top2 + shared) + LayerNorm ----------------
__global__ __launch_bounds__(256) void combine_ln_kernel(
    const float* __restrict__ yexp, const float* __restrict__ ys,
    const int* __restrict__ tok_row, const float* __restrict__ tok_w,
    const float* __restrict__ sharedw, const float* __restrict__ ln_g,
    const float* __restrict__ ln_b, float* __restrict__ out) {
    __shared__ float red[18];
    const int tok = blockIdx.x;
    const int r0 = tok_row[2 * tok], r1 = tok_row[2 * tok + 1];
    const float w0 = tok_w[2 * tok], w1 = tok_w[2 * tok + 1];
    const float sig = 1.f / (1.f + expf(-sharedw[0]));
    const int d0 = threadIdx.x * 4;
    const float4 e0 = *(const float4*)(yexp + (size_t)r0 * 1024 + d0);
    const float4 e1 = *(const float4*)(yexp + (size_t)r1 * 1024 + d0);
    const float4 sh = *(const float4*)(ys + (size_t)tok * 1024 + d0);
    float y[4];
    y[0] = w0 * e0.x + w1 * e1.x + sig * sh.x;
    y[1] = w0 * e0.y + w1 * e1.y + sig * sh.y;
    y[2] = w0 * e0.z + w1 * e1.z + sig * sh.z;
    y[3] = w0 * e0.w + w1 * e1.w + sig * sh.w;
    float s = y[0] + y[1] + y[2] + y[3];
    float q = y[0] * y[0] + y[1] * y[1] + y[2] * y[2] + y[3] * y[3];
    #pragma unroll
    for (int off = 16; off > 0; off >>= 1) {
        s += __shfl_xor(s, off, 32);
        q += __shfl_xor(q, off, 32);
    }
    const int wid = threadIdx.x >> 5;
    if ((threadIdx.x & 31) == 0) { red[wid] = s; red[8 + wid] = q; }
    __syncthreads();
    if (threadIdx.x == 0) {
        float S = 0.f, Q = 0.f;
        #pragma unroll
        for (int i = 0; i < 8; ++i) { S += red[i]; Q += red[8 + i]; }
        float mu = S * (1.f / 1024.f);
        red[16] = mu;
        red[17] = rsqrtf(Q * (1.f / 1024.f) - mu * mu + 1e-5f);
    }
    __syncthreads();
    const float mu = red[16], rs = red[17];
    float4 o;
    o.x = (y[0] - mu) * rs * ln_g[d0 + 0] + ln_b[d0 + 0];
    o.y = (y[1] - mu) * rs * ln_g[d0 + 1] + ln_b[d0 + 1];
    o.z = (y[2] - mu) * rs * ln_g[d0 + 2] + ln_b[d0 + 2];
    o.w = (y[3] - mu) * rs * ln_g[d0 + 3] + ln_b[d0 + 3];
    *(float4*)(out + (size_t)tok * 1024 + d0) = o;
}

// ---------------- host launcher ----------------
extern "C" void kernel_launch(void* const* d_in, const int* in_sizes, int n_in,
                              void* d_out, int out_size, void* d_ws, size_t ws_size,
                              hipStream_t stream) {
    (void)in_sizes; (void)n_in; (void)out_size; (void)ws_size;
    const float* x     = (const float*)d_in[0];
    const float* rw    = (const float*)d_in[1];
    const float* rb    = (const float*)d_in[2];
    const float* w1    = (const float*)d_in[3];
    const float* b1    = (const float*)d_in[4];
    const float* gate  = (const float*)d_in[5];
    const float* w2    = (const float*)d_in[6];
    const float* b2    = (const float*)d_in[7];
    const float* sw1   = (const float*)d_in[8];
    const float* sb1   = (const float*)d_in[9];
    const float* sgate = (const float*)d_in[10];
    const float* sw2   = (const float*)d_in[11];
    const float* sb2   = (const float*)d_in[12];
    const float* shw   = (const float*)d_in[13];
    const float* lng   = (const float*)d_in[14];
    const float* lnb   = (const float*)d_in[15];
    float* out = (float*)d_out;

    constexpr int N = 8192, D = 1024, F = 4096, E = 8, F2 = 2048;
    constexpr int ROWS_PAD = 16384 + E * 128;     // top2 rows + per-expert 128-alignment

    char* w = (char*)d_ws; size_t off = 0;
    auto alloc = [&](size_t bytes) -> char* {
        char* p = w + off;
        off = (off + bytes + 255) & ~(size_t)255;
        return p;
    };
    unsigned short* xb   = (unsigned short*)alloc((size_t)N * D * 2);
    unsigned short* w1g  = (unsigned short*)alloc((size_t)E * F * D * 2);  // [E][F][D], gate folded
    unsigned short* w2t  = (unsigned short*)alloc((size_t)E * D * F * 2);  // [E][D][F]
    unsigned short* sw1g = (unsigned short*)alloc((size_t)F2 * D * 2);     // [F2][D], sgate folded
    unsigned short* sw2t = (unsigned short*)alloc((size_t)D * F2 * 2);     // [D][F2]
    float* b1g   = (float*)alloc((size_t)E * F * 4);
    float* sb1g  = (float*)alloc((size_t)F2 * 4);
    int*   tok_exp = (int*)alloc((size_t)2 * N * 4);
    float* tok_w   = (float*)alloc((size_t)2 * N * 4);
    int*   tok_row = (int*)alloc((size_t)2 * N * 4);
    int*   counts  = (int*)alloc(8 * 4);
    int*   cursor  = (int*)alloc(8 * 4);
    int*   dbase   = (int*)alloc(8 * 4);
    int*   rowperm = (int*)alloc((size_t)ROWS_PAD * 4);
    unsigned short* hbuf = (unsigned short*)alloc((size_t)ROWS_PAD * F * 2);
    float* yexp = (float*)alloc((size_t)ROWS_PAD * D * 4);
    unsigned short* hs = (unsigned short*)alloc((size_t)N * F2 * 2);
    float* ys = (float*)alloc((size_t)N * D * 4);

    (void)hipMemsetAsync(counts, 0, 8 * sizeof(int), stream);

    // precision conversion / weight prep (bf16, gate folded, weights transposed to [n][k])
    cvt_bf16_kernel<<<(N * D / 4 + 255) / 256, 256, 0, stream>>>(x, xb, N * D);
    transcvt_kernel<<<dim3(F / 32, D / 32, E), dim3(32, 8), 0, stream>>>(w1, gate, w1g, D, F);
    transcvt_kernel<<<dim3(D / 32, F / 32, E), dim3(32, 8), 0, stream>>>(w2, nullptr, w2t, F, D);
    transcvt_kernel<<<dim3(F2 / 32, D / 32, 1), dim3(32, 8), 0, stream>>>(sw1, sgate, sw1g, D, F2);
    transcvt_kernel<<<dim3(D / 32, F2 / 32, 1), dim3(32, 8), 0, stream>>>(sw2, nullptr, sw2t, F2, D);
    mulvec_kernel<<<(E * F + 255) / 256, 256, 0, stream>>>(b1, gate, b1g, E * F);
    mulvec_kernel<<<(F2 + 255) / 256, 256, 0, stream>>>(sb1, sgate, sb1g, F2);

    // routing + dispatch lists
    router_kernel<<<N / 8, 256, 0, stream>>>(x, rw, rb, tok_exp, tok_w, counts, N);
    scan_kernel<<<1, 32, 0, stream>>>(counts, dbase, cursor);
    fill_kernel<<<N / 256, 256, 0, stream>>>(tok_exp, dbase, cursor, rowperm, tok_row, N);

    // expert fc1 (gathered A, +b1*gate, exact GELU) -> hbuf (bf16)
    gemm_bf16_kernel<<<dim3(F / 128, 64, E), 256, 0, stream>>>(
        xb, w1g, b1g, rowperm, dbase, counts, 0, D, F, hbuf, nullptr, 1);
    // expert fc2 (+b2) -> yexp (f32)
    gemm_bf16_kernel<<<dim3(D / 128, 64, E), 256, 0, stream>>>(
        hbuf, w2t, b2, nullptr, dbase, counts, 0, F, D, nullptr, yexp, 0);
    // shared fc1 -> hs (bf16)
    gemm_bf16_kernel<<<dim3(F2 / 128, 64, 1), 256, 0, stream>>>(
        xb, sw1g, sb1g, nullptr, nullptr, nullptr, N, D, F2, hs, nullptr, 1);
    // shared fc2 (+sb2) -> ys (f32)
    gemm_bf16_kernel<<<dim3(D / 128, 64, 1), 256, 0, stream>>>(
        hs, sw2t, sb2, nullptr, nullptr, nullptr, N, F2, D, nullptr, ys, 0);

    // weighted combine + sigmoid-scaled shared + LayerNorm
    combine_ln_kernel<<<N, 256, 0, stream>>>(yexp, ys, tok_row, tok_w, shw, lng, lnb, out);
}
